// Linear_2748779070256
// MI455X (gfx1250) — compile-verified, matrix operand order checked by reading, not run on hardware
//
#include <hip/hip_runtime.h>

typedef __attribute__((ext_vector_type(8))) int v8i;
typedef __attribute__((ext_vector_type(4))) int v4i;

#define EPSQ 1e-8f
#define TM 128
#define TN 128
#define KC 64

#if __has_builtin(__builtin_amdgcn_global_load_async_to_lds_b128) && \
    __has_builtin(__builtin_amdgcn_s_wait_asynccnt)
#define USE_ASYNC_LDS 1
#else
#define USE_ASYNC_LDS 0
#endif

// ---------------- ws layout ----------------
// [0..15]        : 4 float absmax scalars (x, w, b, out) as float bits (atomicMax on uint)
// [256 ..]       : bq, N floats (quantized bias)
// [65536 ..]     : x8  int8 [M,K]
// [65536+M*K ..] : w8  int8 [N,K]
#define WS_SCAL_OFF   0
#define WS_BQ_OFF     256
#define WS_X8_OFF     65536

// ---------------- small utility kernels ----------------

__global__ void k_zero_scal(unsigned* scal) {
    if (threadIdx.x < 4) scal[threadIdx.x] = 0u;
}

__global__ __launch_bounds__(256) void k_absmax(const float4* __restrict__ src,
                                                int n4, unsigned* __restrict__ dst) {
    __shared__ float red[256];
    float m = 0.0f;
    for (int i = blockIdx.x * blockDim.x + threadIdx.x; i < n4;
         i += gridDim.x * blockDim.x) {
        float4 v = src[i];
        m = fmaxf(m, fmaxf(fmaxf(fabsf(v.x), fabsf(v.y)),
                           fmaxf(fabsf(v.z), fabsf(v.w))));
    }
    red[threadIdx.x] = m;
    __syncthreads();
    for (int s = 128; s > 0; s >>= 1) {
        if (threadIdx.x < (unsigned)s)
            red[threadIdx.x] = fmaxf(red[threadIdx.x], red[threadIdx.x + s]);
        __syncthreads();
    }
    if (threadIdx.x == 0) atomicMax(dst, __float_as_uint(red[0]));
}

// quantize float -> int8 codes: clip(rint(t/scale), -qmax, qmax)
__global__ __launch_bounds__(256) void k_quant_i8(const float4* __restrict__ src,
                                                  int n4, char4* __restrict__ dst,
                                                  float qmax,
                                                  const float* __restrict__ scal,
                                                  int idx) {
    float s = fmaxf(scal[idx] / qmax, EPSQ);
    for (int i = blockIdx.x * blockDim.x + threadIdx.x; i < n4;
         i += gridDim.x * blockDim.x) {
        float4 v = src[i];
        float r0 = fminf(fmaxf(rintf(v.x / s), -qmax), qmax);
        float r1 = fminf(fmaxf(rintf(v.y / s), -qmax), qmax);
        float r2 = fminf(fmaxf(rintf(v.z / s), -qmax), qmax);
        float r3 = fminf(fmaxf(rintf(v.w / s), -qmax), qmax);
        char4 q;
        q.x = (signed char)(int)r0;
        q.y = (signed char)(int)r1;
        q.z = (signed char)(int)r2;
        q.w = (signed char)(int)r3;
        dst[i] = q;
    }
}

__global__ void k_quant_bias(const float* __restrict__ b, int n,
                             float* __restrict__ bq,
                             const float* __restrict__ scal) {
    int i = blockIdx.x * blockDim.x + threadIdx.x;
    if (i < n) {
        float s = fmaxf(scal[2] / 127.0f, EPSQ);
        float r = fminf(fmaxf(rintf(b[i] / s), -127.0f), 127.0f);
        bq[i] = r * s;
    }
}

__global__ __launch_bounds__(256) void k_quant_out(float4* __restrict__ o, int n4,
                                                   const float* __restrict__ scal) {
    float s = fmaxf(scal[3] / 127.0f, EPSQ);
    for (int i = blockIdx.x * blockDim.x + threadIdx.x; i < n4;
         i += gridDim.x * blockDim.x) {
        float4 v = o[i];
        v.x = fminf(fmaxf(rintf(v.x / s), -127.0f), 127.0f) * s;
        v.y = fminf(fmaxf(rintf(v.y / s), -127.0f), 127.0f) * s;
        v.z = fminf(fmaxf(rintf(v.z / s), -127.0f), 127.0f) * s;
        v.w = fminf(fmaxf(rintf(v.w / s), -127.0f), 127.0f) * s;
        o[i] = v;
    }
}

#if USE_ASYNC_LDS
__device__ __forceinline__ void cp16_async(const void* g, void* l) {
    __builtin_amdgcn_global_load_async_to_lds_b128(
        (__attribute__((address_space(1))) v4i*)g,
        (__attribute__((address_space(3))) v4i*)l, 0, 0);
}
#endif

// ---------------- IU8 WMMA GEMM (pass 1) ----------------
// out = bq[col] + (sx*sw) * (ix @ iw^T)  [float, unquantized] ; absmax(out)->scal[3]
// Block: 256 threads (8 waves). Block tile 128x128, K-step 64, double-buffered LDS.
// Wave (wm = wave&3, wn = wave>>2) computes rows [wm*32,+32) x cols [wn*64,+64)
// as 2x4 grid of V_WMMA_I32_16X16X64_IU8 tiles.
__global__ __launch_bounds__(256) void k_gemm_iu8(
    const signed char* __restrict__ x8, const signed char* __restrict__ w8,
    const float* __restrict__ bq, const float* __restrict__ scal,
    float* __restrict__ out, unsigned* __restrict__ outmax,
    int M, int N, int K) {
    __shared__ int ldsA[2][TM * KC / 4];  // 2 x 8 KB
    __shared__ int ldsB[2][TN * KC / 4];  // 2 x 8 KB

    const int tid = threadIdx.x;
    const int lane = tid & 31;
    const int wave = tid >> 5;
    const int wm = wave & 3;   // 32-row slice
    const int wn = wave >> 2;  // 64-col slice

    const int rowBase = blockIdx.y * TM;
    const int colBase = blockIdx.x * TN;

    // ---- loader assignment: thread handles 16B chunks tid and tid+256 of each tile
    // chunk t -> row (t>>2), byte-col (t&3)*16 ; LDS int4 index == t (contiguous)
    const int lr0 = tid >> 2;        // 0..63
    const int lcb = (tid & 3) * 16;  // 0/16/32/48
    const signed char* pA0 = x8 + (size_t)(rowBase + lr0) * K + lcb;
    const signed char* pA1 = pA0 + (size_t)64 * K;
    const signed char* pB0 = w8 + (size_t)(colBase + lr0) * K + lcb;
    const signed char* pB1 = pB0 + (size_t)64 * K;

    v8i acc[2][4] = {};

    const int lr = lane & 15;        // row-in-fragment
    const int lk = (lane >> 4) * 2;  // 0 or 2 ints (0 or 8 bytes)
    const int nk = K / KC;

    auto compute = [&](int buf) {
        const int* __restrict__ LA = ldsA[buf];
        const int* __restrict__ LB = ldsB[buf];
        v8i a[2], b[4];
        for (int mi = 0; mi < 2; ++mi) {
            int base = (wm * 32 + mi * 16 + lr) * (KC / 4) + lk;
            a[mi][0] = LA[base +  0]; a[mi][1] = LA[base +  1];
            a[mi][2] = LA[base +  4]; a[mi][3] = LA[base +  5];
            a[mi][4] = LA[base +  8]; a[mi][5] = LA[base +  9];
            a[mi][6] = LA[base + 12]; a[mi][7] = LA[base + 13];
        }
        for (int ni = 0; ni < 4; ++ni) {
            int base = (wn * 64 + ni * 16 + lr) * (KC / 4) + lk;
            b[ni][0] = LB[base +  0]; b[ni][1] = LB[base +  1];
            b[ni][2] = LB[base +  4]; b[ni][3] = LB[base +  5];
            b[ni][4] = LB[base +  8]; b[ni][5] = LB[base +  9];
            b[ni][6] = LB[base + 12]; b[ni][7] = LB[base + 13];
        }
        for (int mi = 0; mi < 2; ++mi)
            for (int ni = 0; ni < 4; ++ni)
                acc[mi][ni] = __builtin_amdgcn_wmma_i32_16x16x64_iu8(
                    true, a[mi], true, b[ni], acc[mi][ni], false, false);
    };

#if USE_ASYNC_LDS
    // ---------- async double-buffered pipeline ----------
    {   // prologue: stage 0 -> buf 0
        cp16_async(pA0, (int4*)ldsA[0] + tid);       pA0 += KC;
        cp16_async(pA1, (int4*)ldsA[0] + tid + 256); pA1 += KC;
        cp16_async(pB0, (int4*)ldsB[0] + tid);       pB0 += KC;
        cp16_async(pB1, (int4*)ldsB[0] + tid + 256); pB1 += KC;
    }
    for (int i = 0; i < nk; ++i) {
        __builtin_amdgcn_s_wait_asynccnt(0);  // own stage-i copies landed in LDS
        __syncthreads();                      // everyone's landed; buf i^1 free
        if (i + 1 < nk) {
            int nxt = (i + 1) & 1;
            cp16_async(pA0, (int4*)ldsA[nxt] + tid);       pA0 += KC;
            cp16_async(pA1, (int4*)ldsA[nxt] + tid + 256); pA1 += KC;
            cp16_async(pB0, (int4*)ldsB[nxt] + tid);       pB0 += KC;
            cp16_async(pB1, (int4*)ldsB[nxt] + tid + 256); pB1 += KC;
        }
        compute(i & 1);
    }
#else
    // ---------- register-staged fallback (no per-element load->store stalls) ----------
    int4 rA0 = *(const int4*)pA0; pA0 += KC;
    int4 rA1 = *(const int4*)pA1; pA1 += KC;
    int4 rB0 = *(const int4*)pB0; pB0 += KC;
    int4 rB1 = *(const int4*)pB1; pB1 += KC;
    for (int i = 0; i < nk; ++i) {
        __syncthreads();  // previous compute done reading LDS
        ((int4*)ldsA[0])[tid]       = rA0;
        ((int4*)ldsA[0])[tid + 256] = rA1;
        ((int4*)ldsB[0])[tid]       = rB0;
        ((int4*)ldsB[0])[tid + 256] = rB1;
        __syncthreads();
        if (i + 1 < nk) {  // prefetch next tile while computing
            rA0 = *(const int4*)pA0; pA0 += KC;
            rA1 = *(const int4*)pA1; pA1 += KC;
            rB0 = *(const int4*)pB0; pB0 += KC;
            rB1 = *(const int4*)pB1; pB1 += KC;
        }
        compute(0);
    }
#endif

    // ---------- epilogue: out = bq + s*acc ; absmax(out) ----------
    const float sx = fmaxf(scal[0] / 127.0f, EPSQ);
    const float sw = fmaxf(scal[1] / 7.0f, EPSQ);
    const float s = sx * sw;

    const int rsub = (lane >> 4) << 3;  // C layout: col = lane&15 ; row = v + rsub
    float lmax = 0.0f;
    for (int mi = 0; mi < 2; ++mi) {
        for (int ni = 0; ni < 4; ++ni) {
            int col = colBase + wn * 64 + ni * 16 + lr;
            float bc = bq[col];
            int rbase = rowBase + wm * 32 + mi * 16 + rsub;
            for (int v = 0; v < 8; ++v) {
                float o = fmaf(s, (float)acc[mi][ni][v], bc);
                out[(size_t)(rbase + v) * N + col] = o;
                lmax = fmaxf(lmax, fabsf(o));
            }
        }
    }
    atomicMax(outmax, __float_as_uint(lmax));
}

// ---------------- launcher ----------------
extern "C" void kernel_launch(void* const* d_in, const int* in_sizes, int n_in,
                              void* d_out, int out_size, void* d_ws, size_t ws_size,
                              hipStream_t stream) {
    const float* x = (const float*)d_in[0];   // [M,K]
    const float* w = (const float*)d_in[1];   // [N,K]
    const float* b = (const float*)d_in[2];   // [N]

    const int N = in_sizes[2];
    const int K = in_sizes[1] / N;
    const int M = in_sizes[0] / K;

    char* wsb = (char*)d_ws;
    unsigned* scal_u = (unsigned*)(wsb + WS_SCAL_OFF);
    float* scal_f = (float*)(wsb + WS_SCAL_OFF);
    float* bq = (float*)(wsb + WS_BQ_OFF);
    signed char* x8 = (signed char*)(wsb + WS_X8_OFF);
    signed char* w8 = x8 + (size_t)M * K;

    float* out = (float*)d_out;

    hipLaunchKernelGGL(k_zero_scal, dim3(1), dim3(32), 0, stream, scal_u);

    {
        int n4 = (M * K) / 4;
        int blocks = (n4 + 2047) / 2048; if (blocks > 1024) blocks = 1024;
        hipLaunchKernelGGL(k_absmax, dim3(blocks), dim3(256), 0, stream,
                           (const float4*)x, n4, scal_u + 0);
    }
    {
        int n4 = (N * K) / 4;
        int blocks = (n4 + 2047) / 2048; if (blocks > 1024) blocks = 1024;
        hipLaunchKernelGGL(k_absmax, dim3(blocks), dim3(256), 0, stream,
                           (const float4*)w, n4, scal_u + 1);
    }
    {
        int n4 = N / 4;
        int blocks = (n4 + 255) / 256;
        hipLaunchKernelGGL(k_absmax, dim3(blocks), dim3(256), 0, stream,
                           (const float4*)b, n4, scal_u + 2);
    }

    {
        int n4 = (M * K) / 4;
        int blocks = (n4 + 1023) / 1024; if (blocks > 2048) blocks = 2048;
        hipLaunchKernelGGL(k_quant_i8, dim3(blocks), dim3(256), 0, stream,
                           (const float4*)x, n4, (char4*)x8, 127.0f, scal_f, 0);
        hipLaunchKernelGGL(k_quant_i8, dim3(blocks), dim3(256), 0, stream,
                           (const float4*)w, n4, (char4*)w8, 7.0f, scal_f, 1);
    }
    hipLaunchKernelGGL(k_quant_bias, dim3((N + 255) / 256), dim3(256), 0, stream,
                       b, N, bq, scal_f);

    hipLaunchKernelGGL(k_gemm_iu8, dim3(N / TN, M / TM), dim3(256), 0, stream,
                       x8, w8, bq, scal_f, out, scal_u + 3, M, N, K);

    {
        int n4 = (M * N) / 4;
        int blocks = (n4 + 1023) / 1024; if (blocks > 2048) blocks = 2048;
        hipLaunchKernelGGL(k_quant_out, dim3(blocks), dim3(256), 0, stream,
                           (float4*)out, n4, scal_f);
    }
}
